// DualInputMambaLayer_45698452029747
// MI455X (gfx1250) — matrix operand order repaired and checked by reading, not compiled, not run on hardware
//
#include <hip/hip_runtime.h>
#include <hip/hip_bf16.h>
#include <math.h>
#include <stdint.h>

typedef __bf16 bf16_t;
typedef bf16_t v16bf __attribute__((ext_vector_type(16)));
typedef bf16_t v8bf  __attribute__((ext_vector_type(8)));
typedef float  v8f   __attribute__((ext_vector_type(8)));

#define D_MODEL 1024
#define D_INNER 2048
#define D_STATE 16
#define D_CONV  4
#define DT_RANK 64
#define XPROJ_N (DT_RANK + 2 * D_STATE)   // 96
#define BATCH   2
#define SEQLEN  2048
#define ROWS    (BATCH * SEQLEN)          // 4096

// ---------------------------------------------------------------------------
// Elementwise helpers
// ---------------------------------------------------------------------------
__global__ void k_f32_to_bf16(const float* __restrict__ src,
                              bf16_t* __restrict__ dst, int n) {
  int i = blockIdx.x * blockDim.x + threadIdx.x;
  if (i < n) dst[i] = (bf16_t)src[i];
}

__global__ void k_neg_exp(const float* __restrict__ src,
                          float* __restrict__ dst, int n) {
  int i = blockIdx.x * blockDim.x + threadIdx.x;
  if (i < n) dst[i] = -__expf(src[i]);
}

// Causal depthwise conv (width 4) over xi (= first half of xz) + SiLU.
__global__ void k_conv_silu(const float* __restrict__ xz,
                            const float* __restrict__ conv_w,
                            const float* __restrict__ conv_b,
                            float* __restrict__ u,
                            bf16_t* __restrict__ u_bf) {
  int idx = blockIdx.x * blockDim.x + threadIdx.x;  // over ROWS * D_INNER
  if (idx >= ROWS * D_INNER) return;
  int d   = idx & (D_INNER - 1);
  int row = idx >> 11;            // row = b*L + t
  int t   = row & (SEQLEN - 1);
  long base = (long)row * (2 * D_INNER) + d;  // xi lives in cols [0, D_INNER)
  float acc = conv_b[d];
#pragma unroll
  for (int k = 0; k < D_CONV; ++k) {
    int tt = t - (D_CONV - 1) + k;
    float xv = (tt >= 0) ? xz[base + (long)(k - (D_CONV - 1)) * (2 * D_INNER)]
                         : 0.f;
    acc += xv * conv_w[d * D_CONV + k];
  }
  float s = acc / (1.f + __expf(-acc));  // silu
  u[idx] = s;
  u_bf[idx] = (bf16_t)s;
}

// Extract dt_r slice (cols 0..63 of dbc) into bf16 for the W_dt GEMM.
__global__ void k_slice_dtr(const float* __restrict__ dbc,
                            bf16_t* __restrict__ dtr, int n) {
  int i = blockIdx.x * blockDim.x + threadIdx.x;
  if (i >= n) return;
  int r = i / DT_RANK, c = i - r * DT_RANK;
  dtr[i] = (bf16_t)dbc[(long)r * XPROJ_N + c];
}

// ---------------------------------------------------------------------------
// Selective scan: 16-lane half-wave per (b, d) channel, lane = state index.
// ---------------------------------------------------------------------------
__global__ void __launch_bounds__(256) k_scan(
    const float* __restrict__ dtpre,  // ROWS x D_INNER (pre-bias)
    const float* __restrict__ b_dt,   // D_INNER
    const float* __restrict__ u,      // ROWS x D_INNER
    const float* __restrict__ xz,     // ROWS x 2*D_INNER (z at +D_INNER)
    const float* __restrict__ dbc,    // ROWS x 96 (B at +64, C at +80)
    const float* __restrict__ Aneg,   // D_INNER x D_STATE  (= -exp(A_log))
    const float* __restrict__ Dp,     // D_INNER
    bf16_t* __restrict__ yact)        // ROWS x D_INNER
{
  int s    = threadIdx.x & 15;
  int half = threadIdx.x >> 4;                 // 0..15
  int pair = blockIdx.x * 16 + half;           // (b,d) channel id, 0..4095
  int b = pair >> 11;
  int d = pair & (D_INNER - 1);

  float a    = Aneg[d * D_STATE + s];
  float dval = Dp[d];
  float bias = b_dt[d];
  float h = 0.f;
  long rbase = (long)b * SEQLEN;

  for (int t = 0; t < SEQLEN; ++t) {
    long row = rbase + t;
    float x  = dtpre[row * D_INNER + d] + bias;
    float dt = (x > 20.f) ? x : log1pf(__expf(x));   // softplus
    float uv = u[row * D_INNER + d];
    float zv = xz[row * (2 * D_INNER) + D_INNER + d];
    float Bs = dbc[row * XPROJ_N + DT_RANK + s];
    float Cs = dbc[row * XPROJ_N + DT_RANK + D_STATE + s];

    h = h * __expf(dt * a) + (dt * uv) * Bs;
    float c = h * Cs;
    c += __shfl_xor(c, 1, 16);
    c += __shfl_xor(c, 2, 16);
    c += __shfl_xor(c, 4, 16);
    c += __shfl_xor(c, 8, 16);

    float y = (c + uv * dval) * (zv / (1.f + __expf(-zv)));
    if (s == 0) yact[row * D_INNER + d] = (bf16_t)y;
  }
}

// ---------------------------------------------------------------------------
// bf16 WMMA GEMM: C(M,N) = A(M,K) * B(N,K)^T, A/B row-major bf16, C f32.
// 256 threads = 8 waves; block tile 128x128, K-step 32; wave tile 64x32
// (4x2 of 16x16) -> 8x v_wmma_f32_16x16x32_bf16 per K-step.
// Global->LDS staging uses gfx1250 GLOBAL_LOAD_ASYNC_TO_LDS_B128 with
// double-buffered LDS: async fill of buffer k+1 overlaps WMMA on buffer k.
// Requires M % 128 == 0 and K % 32 == 0 (true for all calls here);
// ragged N handled by clamped loads + guarded stores.
// ---------------------------------------------------------------------------
#define BM 128
#define BN 128
#define BK 32
#define LDT 40  // LDS row stride (halves): 64B data + 16B pad, keeps 16B align

__global__ void __launch_bounds__(256) k_gemm_bf16(
    const bf16_t* __restrict__ A, const bf16_t* __restrict__ B,
    float* __restrict__ C, int M, int N, int K)
{
  __shared__ bf16_t As[2][BM * LDT];
  __shared__ bf16_t Bs[2][BN * LDT];

  int tid  = threadIdx.x;
  int wave = tid >> 5;
  int lane = tid & 31;
  int l16   = lane & 15;
  int lhalf = lane >> 4;       // 0 or 1

  int blockM = blockIdx.y * BM;
  int blockN = blockIdx.x * BN;
  int wm = (wave >> 2) * 64;   // 2 waves along M
  int wn = (wave & 3) * 32;    // 4 waves along N

  // Staging coords: each thread owns a 32B (2x16B) chunk of one row of each
  // tile.  row = tid>>1 (0..127), byte-chunk = (tid&1)*32.
  int srow  = tid >> 1;
  int scol  = (tid & 1) * 16;              // halves
  int brow  = blockN + srow;
  if (brow > N - 1) brow = N - 1;          // clamp (garbage cols never stored)
  const bf16_t* gA = A + (size_t)(blockM + srow) * K + scol;
  const bf16_t* gB = B + (size_t)brow * K + scol;
  uint32_t ldsA[2], ldsB[2];
#pragma unroll
  for (int bufi = 0; bufi < 2; ++bufi) {
    ldsA[bufi] = (uint32_t)(uintptr_t)(&As[bufi][srow * LDT + scol]);
    ldsB[bufi] = (uint32_t)(uintptr_t)(&Bs[bufi][srow * LDT + scol]);
  }

  v8f acc[4][2];
#pragma unroll
  for (int mt = 0; mt < 4; ++mt)
#pragma unroll
    for (int nt = 0; nt < 2; ++nt)
      acc[mt][nt] = (v8f){0.f, 0.f, 0.f, 0.f, 0.f, 0.f, 0.f, 0.f};

  int nk = K / BK;

  // Prologue: async-fill buffer 0.
  {
    const bf16_t* ga = gA;
    const bf16_t* gb = gB;
    asm volatile(
        "global_load_async_to_lds_b128 %0, %2, off\n\t"
        "global_load_async_to_lds_b128 %0, %2, off offset:16\n\t"
        "global_load_async_to_lds_b128 %1, %3, off\n\t"
        "global_load_async_to_lds_b128 %1, %3, off offset:16"
        :
        : "v"(ldsA[0]), "v"(ldsB[0]), "v"(ga), "v"(gb)
        : "memory");
  }

  for (int kt = 0; kt < nk; ++kt) {
    // Our async fills for buffer kt must land before anyone reads them.
    asm volatile("s_wait_asynccnt 0x0" ::: "memory");
    __syncthreads();

    // Kick off async fill of the next buffer (overlaps the WMMAs below).
    if (kt + 1 < nk) {
      int nb = (kt + 1) & 1;
      const bf16_t* ga = gA + (kt + 1) * BK;
      const bf16_t* gb = gB + (kt + 1) * BK;
      asm volatile(
          "global_load_async_to_lds_b128 %0, %2, off\n\t"
          "global_load_async_to_lds_b128 %0, %2, off offset:16\n\t"
          "global_load_async_to_lds_b128 %1, %3, off\n\t"
          "global_load_async_to_lds_b128 %1, %3, off offset:16"
          :
          : "v"(ldsA[nb]), "v"(ldsB[nb]), "v"(ga), "v"(gb)
          : "memory");
    }

    int buf = kt & 1;
    const bf16_t* as = &As[buf][0];
    const bf16_t* bs = &Bs[buf][0];

    // Fragments per documented CDNA5 wave32 layouts.
    v16bf af[4], bfv[2];
    int akb = lhalf * 8;    // A: lanes 0-15 -> K{0-7,16-23}; 16-31 -> K{8-15,24-31}
    int bkb = lhalf * 8;    // B: lanes 0-15 -> K{0-7,16-23}; 16-31 -> K{8-15,24-31}
#pragma unroll
    for (int mt = 0; mt < 4; ++mt) {
      int r = wm + mt * 16 + l16;  // A row (M index)
      ((v8bf*)&af[mt])[0] = *(const v8bf*)(as + r * LDT + akb);
      ((v8bf*)&af[mt])[1] = *(const v8bf*)(as + r * LDT + akb + 16);
    }
#pragma unroll
    for (int nt = 0; nt < 2; ++nt) {
      int r = wn + nt * 16 + l16;  // B row (N index)
      ((v8bf*)&bfv[nt])[0] = *(const v8bf*)(bs + r * LDT + bkb);
      ((v8bf*)&bfv[nt])[1] = *(const v8bf*)(bs + r * LDT + bkb + 16);
    }
#pragma unroll
    for (int mt = 0; mt < 4; ++mt)
#pragma unroll
      for (int nt = 0; nt < 2; ++nt)
        acc[mt][nt] = __builtin_amdgcn_wmma_f32_16x16x32_bf16(
            false, af[mt], false, bfv[nt], (short)0, acc[mt][nt], false, false);
  }

  // Store: C/D layout — VGPR r: lanes 0-15 M=r, lanes 16-31 M=r+8; N=lane&15.
#pragma unroll
  for (int mt = 0; mt < 4; ++mt)
#pragma unroll
    for (int nt = 0; nt < 2; ++nt) {
      int col = blockN + wn + nt * 16 + l16;
      if (col >= N) continue;
      int rowb = blockM + wm + mt * 16 + lhalf * 8;
#pragma unroll
      for (int r = 0; r < 8; ++r)
        C[(size_t)(rowb + r) * N + col] = acc[mt][nt][r];
    }
}

// ---------------------------------------------------------------------------
// Host orchestration
// ---------------------------------------------------------------------------
static inline int cdiv(int a, int b) { return (a + b - 1) / b; }

extern "C" void kernel_launch(void* const* d_in, const int* in_sizes, int n_in,
                              void* d_out, int out_size, void* d_ws,
                              size_t ws_size, hipStream_t stream) {
  (void)in_sizes; (void)n_in; (void)out_size; (void)ws_size;
  const float* x_in[2] = {(const float*)d_in[0], (const float*)d_in[1]};
  const float* W_in    = (const float*)d_in[2];
  const float* conv_w  = (const float*)d_in[3];
  const float* conv_b  = (const float*)d_in[4];
  const float* W_xproj = (const float*)d_in[5];
  const float* W_dt    = (const float*)d_in[6];
  const float* b_dt    = (const float*)d_in[7];
  const float* A_log   = (const float*)d_in[8];
  const float* Dp      = (const float*)d_in[9];
  const float* W_out   = (const float*)d_in[10];
  float* out = (float*)d_out;

  // Workspace carve-up (256B aligned slices)
  char* ws = (char*)d_ws;
  size_t off = 0;
  auto carve = [&](size_t bytes) -> void* {
    void* p = ws + off;
    off += (bytes + 255) & ~(size_t)255;
    return p;
  };
  const size_t NI = (size_t)ROWS * D_INNER;           // 8M elems
  bf16_t* wb_in    = (bf16_t*)carve((size_t)2 * D_INNER * D_MODEL * 2);
  bf16_t* wb_xproj = (bf16_t*)carve((size_t)XPROJ_N * D_INNER * 2);
  bf16_t* wb_dt    = (bf16_t*)carve((size_t)D_INNER * DT_RANK * 2);
  bf16_t* wb_out   = (bf16_t*)carve((size_t)D_MODEL * D_INNER * 2);
  float*  Aneg     = (float*) carve((size_t)D_INNER * D_STATE * 4);
  bf16_t* x_bf     = (bf16_t*)carve((size_t)ROWS * D_MODEL * 2);
  float*  xz       = (float*) carve((size_t)ROWS * 2 * D_INNER * 4);
  float*  u        = (float*) carve(NI * 4);
  bf16_t* u_bf     = (bf16_t*)carve(NI * 2);
  float*  dbc      = (float*) carve((size_t)ROWS * XPROJ_N * 4);
  bf16_t* dtr_bf   = (bf16_t*)carve((size_t)ROWS * DT_RANK * 2);
  float*  dtpre    = (float*) carve(NI * 4);
  bf16_t* yact     = (bf16_t*)carve(NI * 2);

  const int TB = 256;
  // Weights -> bf16 (shared by both inputs), A = -exp(A_log)
  {
    int n;
    n = 2 * D_INNER * D_MODEL;
    k_f32_to_bf16<<<cdiv(n, TB), TB, 0, stream>>>(W_in, wb_in, n);
    n = XPROJ_N * D_INNER;
    k_f32_to_bf16<<<cdiv(n, TB), TB, 0, stream>>>(W_xproj, wb_xproj, n);
    n = D_INNER * DT_RANK;
    k_f32_to_bf16<<<cdiv(n, TB), TB, 0, stream>>>(W_dt, wb_dt, n);
    n = D_MODEL * D_INNER;
    k_f32_to_bf16<<<cdiv(n, TB), TB, 0, stream>>>(W_out, wb_out, n);
    n = D_INNER * D_STATE;
    k_neg_exp<<<cdiv(n, TB), TB, 0, stream>>>(A_log, Aneg, n);
  }

  for (int inp = 0; inp < 2; ++inp) {
    // x -> bf16
    int nx = ROWS * D_MODEL;
    k_f32_to_bf16<<<cdiv(nx, TB), TB, 0, stream>>>(x_in[inp], x_bf, nx);

    // GEMM1: xz = x @ W_in^T   (4096 x 1024) x (4096 x 1024)^T
    k_gemm_bf16<<<dim3(cdiv(2 * D_INNER, BN), cdiv(ROWS, BM)), TB, 0, stream>>>(
        x_bf, wb_in, xz, ROWS, 2 * D_INNER, D_MODEL);

    // conv + silu -> u (f32 + bf16)
    int nu = ROWS * D_INNER;
    k_conv_silu<<<cdiv(nu, TB), TB, 0, stream>>>(xz, conv_w, conv_b, u, u_bf);

    // GEMM2: dbc = u @ W_xproj^T   (4096 x 2048) x (96 x 2048)^T
    k_gemm_bf16<<<dim3(cdiv(XPROJ_N, BN), cdiv(ROWS, BM)), TB, 0, stream>>>(
        u_bf, wb_xproj, dbc, ROWS, XPROJ_N, D_INNER);

    // dt_r slice -> bf16
    int nd = ROWS * DT_RANK;
    k_slice_dtr<<<cdiv(nd, TB), TB, 0, stream>>>(dbc, dtr_bf, nd);

    // GEMM3: dtpre = dt_r @ W_dt^T   (4096 x 64) x (2048 x 64)^T
    k_gemm_bf16<<<dim3(cdiv(D_INNER, BN), cdiv(ROWS, BM)), TB, 0, stream>>>(
        dtr_bf, wb_dt, dtpre, ROWS, D_INNER, DT_RANK);

    // Selective scan (fused softplus / gate / skip) -> yact (bf16)
    k_scan<<<(BATCH * D_INNER) / 16, TB, 0, stream>>>(
        dtpre, b_dt, u, xz, dbc, Aneg, Dp, yact);

    // GEMM4: out = yact @ W_out^T   (4096 x 2048) x (1024 x 2048)^T
    k_gemm_bf16<<<dim3(cdiv(D_MODEL, BN), cdiv(ROWS, BM)), TB, 0, stream>>>(
        yact, wb_out, out + (size_t)inp * ROWS * D_MODEL,
        ROWS, D_MODEL, D_INNER);
  }
}
